// QuantizedLinear_12214886990526
// MI455X (gfx1250) — matrix operand order repaired and checked by reading, not compiled, not run on hardware
//
#include <hip/hip_runtime.h>

typedef __attribute__((ext_vector_type(16))) _Float16 v16h;
typedef __attribute__((ext_vector_type(8)))  float    v8f;

union Frag16 { uint4 u[2]; v16h h; };
union H8     { _Float16 h[8]; uint4 u; };

// Vector-of-4-int types in the address spaces the async-copy builtin expects:
typedef int vi4 __attribute__((vector_size(16)));
typedef __attribute__((address_space(1))) vi4 gvi4;   // global (AS1)
typedef __attribute__((address_space(3))) vi4 lvi4;   // LDS (AS3)

#if __has_builtin(__builtin_amdgcn_global_load_async_to_lds_b128)
#define HAVE_ASYNC_LDS 1
#else
#define HAVE_ASYNC_LDS 0
#endif

#if __has_builtin(__builtin_amdgcn_s_wait_asynccnt)
#define WAIT_ASYNC(n) __builtin_amdgcn_s_wait_asynccnt(n)
#else
#define WAIT_ASYNC(n) asm volatile("s_wait_asynccnt %0" ::"i"(n) : "memory")
#endif

// 16-byte global -> LDS copy (async on gfx1250; sync fallback otherwise)
__device__ __forceinline__ void copy16_to_lds(const _Float16* g, _Float16* l) {
#if HAVE_ASYNC_LDS
    __builtin_amdgcn_global_load_async_to_lds_b128((gvi4*)g, (lvi4*)l,
                                                   /*offset=*/0, /*cpol=*/0);
#else
    *(uint4*)l = *(const uint4*)g;
#endif
}

// ---------------- Prologue kernels: build f16 operands in workspace ----------------

__global__ __launch_bounds__(256)
void x_to_f16_kernel(const float* __restrict__ x, _Float16* __restrict__ x16, long long n8) {
    long long i = (long long)blockIdx.x * blockDim.x + threadIdx.x;
    if (i >= n8) return;
    const float4* p = (const float4*)(x + i * 8);
    float4 a = p[0], b = p[1];
    H8 o;
    o.h[0] = (_Float16)a.x; o.h[1] = (_Float16)a.y;
    o.h[2] = (_Float16)a.z; o.h[3] = (_Float16)a.w;
    o.h[4] = (_Float16)b.x; o.h[5] = (_Float16)b.y;
    o.h[6] = (_Float16)b.z; o.h[7] = (_Float16)b.w;
    *(uint4*)(x16 + i * 8) = o.u;
}

__global__ __launch_bounds__(256)
void dequant_w_kernel(const int* __restrict__ wq, const float* __restrict__ sc,
                      const float* __restrict__ zp, _Float16* __restrict__ w16,
                      int K, int ngroups, long long n8) {
    long long i = (long long)blockIdx.x * blockDim.x + threadIdx.x;
    if (i >= n8) return;
    int per_row = K >> 3;
    int n  = (int)(i / per_row);
    int k0 = (int)(i % per_row) << 3;
    int g  = k0 / (K / ngroups);
    float s = sc[(long long)n * ngroups + g];
    float z = zp[(long long)n * ngroups + g];
    const int4* p = (const int4*)(wq + (long long)n * K + k0);
    int4 a = p[0], b = p[1];
    H8 o;
    o.h[0] = (_Float16)(((float)a.x - z) * s);
    o.h[1] = (_Float16)(((float)a.y - z) * s);
    o.h[2] = (_Float16)(((float)a.z - z) * s);
    o.h[3] = (_Float16)(((float)a.w - z) * s);
    o.h[4] = (_Float16)(((float)b.x - z) * s);
    o.h[5] = (_Float16)(((float)b.y - z) * s);
    o.h[6] = (_Float16)(((float)b.z - z) * s);
    o.h[7] = (_Float16)(((float)b.w - z) * s);
    *(uint4*)(w16 + (long long)n * K + k0) = o.u;
}

// ---------------- f16 WMMA GEMM: C[M,N] = A[M,K] * Bw[N,K]^T + bias ----------------
// Block: 256 threads = 8 wave32, tile 256(M) x 128(N), BK=64, double-buffered LDS.
// Wave tile: 64x64 = 4x4 WMMA accumulators (128 VGPRs) — 16 ds_load_b128 per
// 16 v_wmma per K=32 step.

#define BM 256
#define BN 128
#define BK 64
#define LDA 72   // padded LDS row stride (halves)
#define LDB 72

__global__ __launch_bounds__(256)
void gemm_f16_wmma_kernel(const _Float16* __restrict__ A,   // M x K row-major
                          const _Float16* __restrict__ Bw,  // N x K row-major
                          const float* __restrict__ bias,
                          float* __restrict__ C,
                          int M, int N, int K) {
    __shared__ __align__(16) _Float16 As[2][BM * LDA];   // 2 x 36 KB
    __shared__ __align__(16) _Float16 Bs[2][BN * LDB];   // 2 x 18 KB

    const int tid  = threadIdx.x;
    const int wid  = tid >> 5;
    const int lane = tid & 31;
    const int wm   = wid >> 1;       // 0..3 : M offset wm*64
    const int wn   = wid & 1;        // 0..1 : N offset wn*64
    const int l16  = lane & 15;
    const int hi16 = (lane >> 4) & 1;

    const int block_m = blockIdx.y * BM;
    const int block_n = blockIdx.x * BN;

    v8f acc[4][4];
#pragma unroll
    for (int mt = 0; mt < 4; ++mt)
#pragma unroll
        for (int nt = 0; nt < 4; ++nt)
#pragma unroll
            for (int r = 0; r < 8; ++r)
                acc[mt][nt][r] = 0.0f;

    // Stage one K-tile (A: 256x64 halves, B: 128x64 halves) into buffer `buf`.
    // Per wave: 8 + 4 = 12 async-copy instructions (ASYNCcnt += 12).
    auto stage = [&](int buf, int k0) {
#pragma unroll
        for (int i = 0; i < 8; ++i) {
            int idx = tid + i * 256;          // 0..2047
            int r   = idx >> 3;               // 0..255
            int c8  = (idx & 7) << 3;
            copy16_to_lds(A + (long long)(block_m + r) * K + k0 + c8,
                          &As[buf][r * LDA + c8]);
        }
#pragma unroll
        for (int i = 0; i < 4; ++i) {
            int idx = tid + i * 256;          // 0..1023
            int r   = idx >> 3;               // 0..127
            int c8  = (idx & 7) << 3;
            copy16_to_lds(Bw + (long long)(block_n + r) * K + k0 + c8,
                          &Bs[buf][r * LDB + c8]);
        }
    };

    stage(0, 0);

    for (int k0 = 0; k0 < K; k0 += BK) {
        const int buf = (k0 / BK) & 1;

        if (k0 + BK < K) {
            stage(buf ^ 1, k0 + BK);   // prefetch next tile into other buffer
            WAIT_ASYNC(12);            // current tile's 12 async ops done (in-order)
        } else {
            WAIT_ASYNC(0);
        }
        __syncthreads();               // current tile visible to all waves

#pragma unroll
        for (int kk = 0; kk < BK; kk += 32) {
            // A fragments (ISA 16-bit A 16x32 layout)
            v16h afrag[4];
#pragma unroll
            for (int mt = 0; mt < 4; ++mt) {
                int m   = wm * 64 + mt * 16 + l16;
                int klo = kk + (hi16 ? 8 : 0);
                Frag16 f;
                f.u[0] = *(const uint4*)(&As[buf][m * LDA + klo]);
                f.u[1] = *(const uint4*)(&As[buf][m * LDA + klo + 16]);
                afrag[mt] = f.h;
            }
            // B fragments (32x16: lanes 0-15 K=0..7,16..23, lanes 16-31 K=8..15,24..31)
#pragma unroll
            for (int nt = 0; nt < 4; ++nt) {
                int n  = wn * 64 + nt * 16 + l16;
                int kb = kk + (hi16 ? 8 : 0);
                Frag16 f;
                f.u[0] = *(const uint4*)(&Bs[buf][n * LDB + kb]);
                f.u[1] = *(const uint4*)(&Bs[buf][n * LDB + kb + 16]);
                v16h bfrag = f.h;
#pragma unroll
                for (int mt = 0; mt < 4; ++mt) {
                    acc[mt][nt] = __builtin_amdgcn_wmma_f32_16x16x32_f16(
                        false, afrag[mt], false, bfrag,
                        (short)0, acc[mt][nt], false, false);
                }
            }
        }
        __syncthreads();               // all waves done reading buf before reuse
    }

    // Store with fused bias (C layout: lanes 0-15 M=vgpr, lanes 16-31 M=vgpr+8)
#pragma unroll
    for (int nt = 0; nt < 4; ++nt) {
        int n = block_n + wn * 64 + nt * 16 + l16;
        float bv = bias[n];
#pragma unroll
        for (int mt = 0; mt < 4; ++mt) {
            int mbase = block_m + wm * 64 + mt * 16 + (hi16 ? 8 : 0);
#pragma unroll
            for (int r = 0; r < 8; ++r) {
                C[(long long)(mbase + r) * N + n] = acc[mt][nt][r] + bv;
            }
        }
    }
}

// ---------------- Launch ----------------

extern "C" void kernel_launch(void* const* d_in, const int* in_sizes, int n_in,
                              void* d_out, int out_size, void* d_ws, size_t ws_size,
                              hipStream_t stream) {
    const float* x    = (const float*)d_in[0];
    const int*   wq   = (const int*)d_in[1];
    const float* sc   = (const float*)d_in[2];
    const float* zp   = (const float*)d_in[3];
    const float* bias = (const float*)d_in[4];
    float*       out  = (float*)d_out;

    const int K = 4096;                       // D_IN
    const int N = in_sizes[4];                // D_OUT (bias length)
    const int M = in_sizes[0] / K;            // B*S
    const int ngroups = in_sizes[2] / N;      // 32

    _Float16* x16 = (_Float16*)d_ws;
    _Float16* w16 = x16 + (long long)M * K;

    long long nx8 = (long long)M * K / 8;
    long long nw8 = (long long)N * K / 8;
    int cvt_blocks = (int)((nx8 + 255) / 256);
    int dq_blocks  = (int)((nw8 + 255) / 256);

    x_to_f16_kernel<<<cvt_blocks, 256, 0, stream>>>(x, x16, nx8);
    dequant_w_kernel<<<dq_blocks, 256, 0, stream>>>(wq, sc, zp, w16, K, ngroups, nw8);

    dim3 grid(N / BN, M / BM);
    gemm_f16_wmma_kernel<<<grid, 256, 0, stream>>>(x16, w16, bias, out, M, N, K);
}
